// S4DNeuroModel_85495618995019
// MI455X (gfx1250) — matrix operand order repaired and checked by reading, not compiled, not run on hardware
//
#include <hip/hip_runtime.h>

// ---------------------------------------------------------------------------
// S4D stack for MI455X (gfx1250, wave32).
//  * FFT conv replaced by the mathematically-identical diagonal SSM scan.
//  * All GEMMs in bf16 -> f32 via v_wmma_f32_16x16x32_bf16.
//  * Activations kept bf16 (B,H,L) to halve HBM traffic (23.3 TB/s bound).
//  * GEMM A-tiles staged via Tensor Data Mover (tensor_load_to_lds),
//    double-buffered so the DMA overlaps the WMMA work.
// ---------------------------------------------------------------------------

#define BATCH 16
#define LSEQ  2048
#define DIN   96
#define HCH   512
#define NSTATE 32
#define NLAYER 4

typedef unsigned short u16;
typedef __attribute__((ext_vector_type(16))) __bf16 bf16x16;
typedef __attribute__((ext_vector_type(8)))  float  floatx8;
typedef __attribute__((ext_vector_type(4)))  unsigned int uint32x4;
typedef __attribute__((ext_vector_type(8)))  int int32x8;
typedef __attribute__((ext_vector_type(4)))  int int32x4;

struct alignas(16) U4 { unsigned int a, b, c, d; };
union AFrag { bf16x16 v; U4 q[2]; };

__device__ __forceinline__ float bf2f(u16 h) {
    unsigned int u = ((unsigned int)h) << 16;
    float f; __builtin_memcpy(&f, &u, 4); return f;
}
__device__ __forceinline__ u16 f2bf(float f) {
    unsigned int u; __builtin_memcpy(&u, &f, 4);
    u = (u + 0x7FFFu + ((u >> 16) & 1u)) >> 16;
    return (u16)u;
}

#if __has_builtin(__builtin_amdgcn_tensor_load_to_lds) && \
    __has_builtin(__builtin_amdgcn_s_wait_tensorcnt)
#define USE_TDM 1
#endif

typedef __attribute__((address_space(3))) const void lds_cvoid;
__device__ __forceinline__ unsigned lds_byte_off(const void* p) {
    return (unsigned)(size_t)(lds_cvoid*)p;
}

#ifdef USE_TDM
// ---------------------------------------------------------------------------
// Issue one TDM tile load: global (bf16, 2B elems) -> LDS, contiguous fill.
//   tile0 = contiguous elems (x), tile1 = rows (y, stride0), tile2 = planes
//   (z, stride1). Descriptor layout per CDNA5 ISA.
// This toolchain's builtin is the 6-arg form:
//   (uint32x4 g0, int32x8 g1, int32x4 g2, int32x4 g3, int32x8 extra, i32 cpol)
// ---------------------------------------------------------------------------
__device__ __forceinline__ void tdm_load_tile(const void* gaddr, unsigned lds_off,
                                              int tile0, int tile1, int tile2,
                                              long stride0, long stride1) {
    unsigned long long ga = (unsigned long long)(size_t)gaddr;
    uint32x4 g0;
    g0[0] = 1u;                                    // count=1, is_restore=0
    g0[1] = lds_off;                               // lds_addr [63:32]
    g0[2] = (unsigned)ga;                          // global_addr lo
    g0[3] = (unsigned)(ga >> 32) | (2u << 30);     // global_addr hi | type=2
    const unsigned td0 = 1u << 20, td1 = 1u << 20; // generous OOB bounds
    unsigned long long s0 = (unsigned long long)stride0;
    unsigned long long s1 = (unsigned long long)stride1;
    int32x8 g1;
    g1[0] = (int)(1u << 16);                       // data_size=1 (2 bytes)
    g1[1] = (int)((td0 & 0xFFFFu) << 16);          // tensor_dim0 [79:48]
    g1[2] = (int)((td0 >> 16) | ((td1 & 0xFFFFu) << 16));
    g1[3] = (int)((td1 >> 16) | ((unsigned)tile0 << 16));  // tile_dim0 [127:112]
    g1[4] = (int)((unsigned)tile1 | ((unsigned)tile2 << 16));
    g1[5] = (int)(unsigned)s0;                     // tensor_dim0_stride [207:160]
    g1[6] = (int)((unsigned)(s0 >> 32) | (((unsigned)s1 & 0xFFFFu) << 16));
    g1[7] = (int)(unsigned)(s1 >> 16);             // tensor_dim1_stride [255:208]
    int32x4 g2; g2[0] = (int)(1u << 20); g2[1] = 0; g2[2] = 0; g2[3] = 0;
    int32x4 g3; g3[0] = 0; g3[1] = 0; g3[2] = 0; g3[3] = 0;
    int32x8 g4;
    g4[0] = 0; g4[1] = 0; g4[2] = 0; g4[3] = 0;
    g4[4] = 0; g4[5] = 0; g4[6] = 0; g4[7] = 0;
    __builtin_amdgcn_tensor_load_to_lds(g0, g1, g2, g3, g4, 0);
}
#endif

// ---------------------------------------------------------------------------
// f32 -> bf16 weight conversion
// ---------------------------------------------------------------------------
__global__ void __launch_bounds__(256) cvt_bf16_kernel(const float* __restrict__ s,
                                                       u16* __restrict__ d, int n) {
    int i = blockIdx.x * 256 + threadIdx.x;
    if (i < n) d[i] = f2bf(s[i]);
}

// ---------------------------------------------------------------------------
// Per-(layer,h,n) SSM discretization:
//   dt = exp(log_dt); A = -exp(log_A_real) + i*A_imag; w = exp(dt*A)
//   Cd = (C_re + i C_im) * (w - 1) / A      (store 2*Cd; y uses 2*Re(Cd x))
// ---------------------------------------------------------------------------
__global__ void __launch_bounds__(256) param_prep_kernel(
    const float* __restrict__ log_dt, const float* __restrict__ log_A_real,
    const float* __restrict__ A_imag, const float* __restrict__ C_re,
    const float* __restrict__ C_im, float4* __restrict__ out) {
    int tid = blockIdx.x * 256 + threadIdx.x;      // NLAYER*HCH*NSTATE
    int layer_h = tid / NSTATE;
    float dt  = __expf(log_dt[layer_h]);
    float Are = -__expf(log_A_real[tid]);
    float Aim = A_imag[tid];
    float dre = Are * dt, dim_ = Aim * dt;
    float er  = __expf(dre);
    float wre = er * __cosf(dim_), wim = er * __sinf(dim_);
    float nre = wre - 1.f, nim = wim;              // w - 1
    float cr = C_re[tid], ci = C_im[tid];
    float tre = cr * nre - ci * nim;               // C*(w-1)
    float tim = cr * nim + ci * nre;
    float inv = 1.f / (Are * Are + Aim * Aim);
    float cdre = (tre * Are + tim * Aim) * inv;    // * conj(A)/|A|^2
    float cdim = (tim * Are - tre * Aim) * inv;
    out[tid] = make_float4(wre, wim, 2.f * cdre, 2.f * cdim);
}

// ---------------------------------------------------------------------------
// Input projection: act[b,h,l] = sum_d in[b,l,d]*W_in[h,d] + b_in[h] (bf16 out)
// Block: 64 h x 128 l; 8 waves, each 1 h-tile x 4 l-tiles. K = 96 (3x32).
// A tile staged by TDM (2D), B tile converted f32->bf16 in registers.
// ---------------------------------------------------------------------------
__global__ void __launch_bounds__(256) inproj_kernel(
    const float* __restrict__ in, const u16* __restrict__ Wb,
    const float* __restrict__ bias, u16* __restrict__ act) {
    const int t = threadIdx.x;
    const int lane = t & 31, wave = t >> 5;
    const int wo = wave & 3, wl = wave >> 2;
    const int l0 = blockIdx.x * 128;
    const int h0 = blockIdx.y * 64;
    const int bb = blockIdx.z;
    __shared__ __align__(16) u16 As[64][32];
    __shared__ __align__(16) u16 Bs[128][32];
    floatx8 acc[4] = {};

    for (int kk = 0; kk < DIN; kk += 32) {
        // stage A: W_in rows (K contiguous)
#ifdef USE_TDM
        if (wave == 0)
            tdm_load_tile(Wb + (size_t)h0 * DIN + kk, lds_byte_off(&As[0][0]),
                          32, 64, 1, DIN, 0);
#else
        {
            int row = t >> 2, seg = t & 3;
            const U4* src = reinterpret_cast<const U4*>(Wb + (size_t)(h0 + row) * DIN + kk) + seg;
            *reinterpret_cast<U4*>(&As[row][seg * 8]) = *src;
        }
#endif
        {   // stage B transposed: Bs[n][k] = in[b, l0+n, kk+k]  (f32 -> bf16)
            int n = t >> 1, seg = t & 1;
            const float* src = in + ((size_t)bb * LSEQ + l0 + n) * DIN + kk + seg * 16;
            alignas(16) u16 tmp[16];
#pragma unroll
            for (int j = 0; j < 16; ++j) tmp[j] = f2bf(src[j]);
            *reinterpret_cast<U4*>(&Bs[n][seg * 16])     = *reinterpret_cast<U4*>(tmp);
            *reinterpret_cast<U4*>(&Bs[n][seg * 16 + 8]) = *reinterpret_cast<U4*>(tmp + 8);
        }
#ifdef USE_TDM
        __builtin_amdgcn_s_wait_tensorcnt(0);
#endif
        __syncthreads();
        AFrag a;
        {   // A frag: lanes 0-15 M=0..15 K={kb..kb+7, kb+16..kb+23}, kb = 8*(lane>=16)
            int m = lane & 15, kb = (lane >> 4) * 8;
            a.q[0] = *reinterpret_cast<const U4*>(&As[wo * 16 + m][kb]);
            a.q[1] = *reinterpret_cast<const U4*>(&As[wo * 16 + m][kb + 16]);
        }
#pragma unroll
        for (int lt = 0; lt < 4; ++lt) {
            AFrag bfr;
            int n = lane & 15, kb = (lane >> 4) * 8;
            int row = (wl * 4 + lt) * 16 + n;
            bfr.q[0] = *reinterpret_cast<const U4*>(&Bs[row][kb]);
            bfr.q[1] = *reinterpret_cast<const U4*>(&Bs[row][kb + 16]);
            acc[lt] = __builtin_amdgcn_wmma_f32_16x16x32_bf16(
                false, a.v, false, bfr.v, (short)0, acc[lt], false, false);
        }
        __syncthreads();
    }
#pragma unroll
    for (int lt = 0; lt < 4; ++lt)
#pragma unroll
        for (int e = 0; e < 8; ++e) {
            int m = (lane >> 4) * 8 + e, n = lane & 15;
            int h = h0 + wo * 16 + m;
            int l = l0 + (wl * 4 + lt) * 16 + n;
            act[((size_t)(bb * HCH + h)) * LSEQ + l] = f2bf(acc[lt][e] + bias[h]);
        }
}

// ---------------------------------------------------------------------------
// SSM scan (replaces FFT conv, identical math) + skip + GELU.
// One wave32 per (b,h); lane n holds complex state x_n. Per step:
//   x = w*x + u[t];  y[t] = 2*Re(sum_n Cd_n x_n) + D_h*u[t];  gelu -> bf16.
// ---------------------------------------------------------------------------
__global__ void __launch_bounds__(256) scan_kernel(
    const u16* __restrict__ u, const float4* __restrict__ prm,
    const float* __restrict__ Dvec, u16* __restrict__ y) {
    const int lane = threadIdx.x & 31;
    const int wid = (blockIdx.x * blockDim.x + threadIdx.x) >> 5;  // B*H waves
    const int b = wid >> 9;          // / HCH
    const int h = wid & (HCH - 1);
    float4 p = prm[h * NSTATE + lane];
    const float wre = p.x, wim = p.y, cre = p.z, cim = p.w;
    const float Dh = Dvec[h];
    const u16* ub = u + ((size_t)(b * HCH + h)) * LSEQ;
    u16* yb = y + ((size_t)(b * HCH + h)) * LSEQ;
    float xre = 0.f, xim = 0.f;

    for (int t0 = 0; t0 < LSEQ; t0 += 32) {
        float uch = bf2f(ub[t0 + lane]);          // coalesced 32-step chunk
        float ymine = 0.f;
#pragma unroll
        for (int j = 0; j < 32; ++j) {
            float ut = __shfl(uch, j, 32);
            float nr = fmaf(xre, wre, fmaf(-xim, wim, ut));
            float ni = fmaf(xre, wim, xim * wre);
            xre = nr; xim = ni;
            float s = cre * xre - cim * xim;      // Re(2Cd * x) (2 folded into Cd)
#pragma unroll
            for (int off = 16; off; off >>= 1) s += __shfl_xor(s, off, 32);
            if (lane == j) ymine = fmaf(Dh, ut, s);
        }
        float v = ymine;                           // GELU (tanh approx)
        float g = 0.5f * v * (1.f + tanhf(0.7978845608f * (v + 0.044715f * v * v * v)));
        yb[t0 + lane] = f2bf(g);
    }
}

// ---------------------------------------------------------------------------
// GLU 1x1 conv: z = W_glu(2H x H) @ y(b: H x L) + b_glu; act = z_a*sigmoid(z_b).
// Block: 64 o x 128 l, both GLU halves -> fused epilogue, bf16 out.
// Double-buffered LDS; A tile = 3D TDM load (32k x 64rows x 2 halves) that
// overlaps the WMMA work on the current tile.
// ---------------------------------------------------------------------------
__global__ void __launch_bounds__(256) glu_gemm_kernel(
    const u16* __restrict__ Y, const u16* __restrict__ Wg,
    const float* __restrict__ bglu, u16* __restrict__ act) {
    const int t = threadIdx.x;
    const int lane = t & 31, wave = t >> 5;
    const int wo = wave & 3, wl = wave >> 2;
    const int l0 = blockIdx.x * 128;
    const int h0 = blockIdx.y * 64;
    const int bb = blockIdx.z;
    __shared__ __align__(16) u16 As[2][128][32];  // rows 0..63 half a, 64..127 half b
    __shared__ __align__(16) u16 Bs[2][128][32];  // [n][k] (transposed staging)
    floatx8 acc0[4] = {}, acc1[4] = {};

    auto stageA = [&](int kk, int buf) {
#ifdef USE_TDM
        if (wave == 0)      // z dim covers the two GLU halves (row H+h0 block)
            tdm_load_tile(Wg + (size_t)h0 * HCH + kk, lds_byte_off(&As[buf][0][0]),
                          32, 64, 2, HCH, (long)HCH * HCH);
#else
#pragma unroll
        for (int rep = 0; rep < 2; ++rep) {
            int row = rep * 64 + (t >> 2), seg = t & 3;
            int o = (row < 64) ? (h0 + row) : (HCH + h0 + (row - 64));
            const U4* src = reinterpret_cast<const U4*>(Wg + (size_t)o * HCH + kk) + seg;
            *reinterpret_cast<U4*>(&As[buf][row][seg * 8]) = *src;
        }
#endif
    };
    auto stageB = [&](int kk, int buf) {
        int k = t >> 3, nseg = t & 7;
        const u16* src = Y + ((size_t)(bb * HCH + kk + k)) * LSEQ + l0 + nseg * 16;
        if (kk + 32 < HCH) __builtin_prefetch(src + 32 * LSEQ, 0, 3);
        alignas(16) u16 tmp[16];
        *reinterpret_cast<U4*>(tmp)     = *reinterpret_cast<const U4*>(src);
        *reinterpret_cast<U4*>(tmp + 8) = *reinterpret_cast<const U4*>(src + 8);
#pragma unroll
        for (int j = 0; j < 16; ++j) Bs[buf][nseg * 16 + j][k] = tmp[j];
    };

    // prologue: tile 0
    stageA(0, 0);
    stageB(0, 0);
#ifdef USE_TDM
    __builtin_amdgcn_s_wait_tensorcnt(0);
#endif
    __syncthreads();

    const int KT = HCH / 32;
    for (int kt = 0; kt < KT; ++kt) {
        const int buf = kt & 1;
        if (kt + 1 < KT) stageA((kt + 1) * 32, buf ^ 1);   // DMA overlaps WMMA

        AFrag a0, a1;
        {
            int m = lane & 15, kb = (lane >> 4) * 8;
            a0.q[0] = *reinterpret_cast<const U4*>(&As[buf][wo * 16 + m][kb]);
            a0.q[1] = *reinterpret_cast<const U4*>(&As[buf][wo * 16 + m][kb + 16]);
            a1.q[0] = *reinterpret_cast<const U4*>(&As[buf][64 + wo * 16 + m][kb]);
            a1.q[1] = *reinterpret_cast<const U4*>(&As[buf][64 + wo * 16 + m][kb + 16]);
        }
#pragma unroll
        for (int lt = 0; lt < 4; ++lt) {
            AFrag bfr;
            int n = lane & 15, kb = (lane >> 4) * 8;
            int row = (wl * 4 + lt) * 16 + n;
            bfr.q[0] = *reinterpret_cast<const U4*>(&Bs[buf][row][kb]);
            bfr.q[1] = *reinterpret_cast<const U4*>(&Bs[buf][row][kb + 16]);
            acc0[lt] = __builtin_amdgcn_wmma_f32_16x16x32_bf16(
                false, a0.v, false, bfr.v, (short)0, acc0[lt], false, false);
            acc1[lt] = __builtin_amdgcn_wmma_f32_16x16x32_bf16(
                false, a1.v, false, bfr.v, (short)0, acc1[lt], false, false);
        }
        if (kt + 1 < KT) stageB((kt + 1) * 32, buf ^ 1);
#ifdef USE_TDM
        __builtin_amdgcn_s_wait_tensorcnt(0);   // no-op for non-issuing waves
#endif
        __syncthreads();
    }

#pragma unroll
    for (int lt = 0; lt < 4; ++lt)
#pragma unroll
        for (int e = 0; e < 8; ++e) {
            int m = (lane >> 4) * 8 + e, n = lane & 15;
            int o = h0 + wo * 16 + m;
            int l = l0 + (wl * 4 + lt) * 16 + n;
            float za = acc0[lt][e] + bglu[o];
            float zb = acc1[lt][e] + bglu[HCH + o];
            float g = za / (1.f + __expf(-zb));           // a * sigmoid(b)
            act[((size_t)(bb * HCH + o)) * LSEQ + l] = f2bf(g);
        }
}

// ---------------------------------------------------------------------------
// Output projection: out[b,l,o] = sum_h act[b,h,l]*W_out[o,h] + b_out (D_out=2)
// ---------------------------------------------------------------------------
__global__ void __launch_bounds__(256) outproj_kernel(
    const u16* __restrict__ act, const float* __restrict__ Wout,
    const float* __restrict__ bout, float* __restrict__ out) {
    __shared__ float Ws[2][HCH];
    for (int i = threadIdx.x; i < 2 * HCH; i += 256) Ws[i >> 9][i & (HCH - 1)] = Wout[i];
    __syncthreads();
    int idx = blockIdx.x * 256 + threadIdx.x;      // B*L
    int b = idx >> 11, l = idx & (LSEQ - 1);
    const u16* ab = act + ((size_t)b * HCH) * LSEQ + l;
    float a0 = 0.f, a1 = 0.f;
    for (int h = 0; h < HCH; ++h) {
        float v = bf2f(ab[(size_t)h * LSEQ]);      // coalesced across lanes (l)
        a0 = fmaf(v, Ws[0][h], a0);
        a1 = fmaf(v, Ws[1][h], a1);
    }
    out[idx * 2 + 0] = a0 + bout[0];
    out[idx * 2 + 1] = a1 + bout[1];
}

// ---------------------------------------------------------------------------
// Driver
// ---------------------------------------------------------------------------
extern "C" void kernel_launch(void* const* d_in, const int* in_sizes, int n_in,
                              void* d_out, int out_size, void* d_ws, size_t ws_size,
                              hipStream_t stream) {
    const float* neural_input = (const float*)d_in[0];
    const float* W_in   = (const float*)d_in[1];
    const float* b_in   = (const float*)d_in[2];
    const float* log_dt = (const float*)d_in[3];
    const float* log_A  = (const float*)d_in[4];
    const float* A_imag = (const float*)d_in[5];
    const float* C_re   = (const float*)d_in[6];
    const float* C_im   = (const float*)d_in[7];
    const float* Dvec   = (const float*)d_in[8];
    const float* W_glu  = (const float*)d_in[9];
    const float* b_glu  = (const float*)d_in[10];
    const float* W_out  = (const float*)d_in[11];
    const float* b_out  = (const float*)d_in[12];

    char* ws = (char*)d_ws;
    constexpr size_t SZ_ACT = (size_t)BATCH * HCH * LSEQ * 2;      // 32 MB bf16
    constexpr size_t SZ_WG  = (size_t)NLAYER * 2 * HCH * HCH * 2;  // 4 MB
    constexpr size_t SZ_WI  = (size_t)HCH * DIN * 2;               // 96 KB
    u16*    act    = (u16*)(ws);
    u16*    ybuf   = (u16*)(ws + SZ_ACT);
    u16*    wgbf   = (u16*)(ws + 2 * SZ_ACT);
    u16*    wibf   = (u16*)(ws + 2 * SZ_ACT + SZ_WG);
    float4* params = (float4*)(ws + 2 * SZ_ACT + SZ_WG + SZ_WI);   // 1 MB

    // Precompute: bf16 weights + discretized SSM params
    cvt_bf16_kernel<<<(HCH * DIN + 255) / 256, 256, 0, stream>>>(W_in, wibf, HCH * DIN);
    cvt_bf16_kernel<<<(NLAYER * 2 * HCH * HCH + 255) / 256, 256, 0, stream>>>(
        W_glu, wgbf, NLAYER * 2 * HCH * HCH);
    param_prep_kernel<<<(NLAYER * HCH * NSTATE) / 256, 256, 0, stream>>>(
        log_dt, log_A, A_imag, C_re, C_im, params);

    // Input projection (WMMA bf16 GEMM) -> act (B,H,L) bf16
    inproj_kernel<<<dim3(LSEQ / 128, HCH / 64, BATCH), 256, 0, stream>>>(
        neural_input, wibf, b_in, act);

    // 4 S4D layers: scan(+skip+GELU) -> WMMA GLU GEMM
    for (int layer = 0; layer < NLAYER; ++layer) {
        scan_kernel<<<(BATCH * HCH) / 8, 256, 0, stream>>>(
            act, params + (size_t)layer * HCH * NSTATE, Dvec + layer * HCH, ybuf);
        glu_gemm_kernel<<<dim3(LSEQ / 128, HCH / 64, BATCH), 256, 0, stream>>>(
            ybuf, wgbf + (size_t)layer * 2 * HCH * HCH, b_glu + layer * 2 * HCH, act);
    }

    // Output projection -> f32 (B,L,2)
    outproj_kernel<<<(BATCH * LSEQ) / 256, 256, 0, stream>>>(
        act, W_out, b_out, (float*)d_out);
}
